// MultiHeadAttention_43112881717269
// MI455X (gfx1250) — matrix operand order repaired and checked by reading, not compiled, not run on hardware
//
#include <hip/hip_runtime.h>
#include <hip/hip_bf16.h>

// ---------------------------------------------------------------------------
// MHA forward for MI455X (gfx1250, wave32, WMMA + TDM).
// Shapes: B=2, S=2048, D=1024, H=16, Kd=64.  M = B*S = 4096.
// d_out = [ out (B,S,D) f32 | attn (B,H,S,S) f32 ].
// ---------------------------------------------------------------------------

typedef __bf16 bf16;
typedef __attribute__((ext_vector_type(4)))  bf16  bf16x4;
typedef __attribute__((ext_vector_type(8)))  bf16  bf16x8;
typedef __attribute__((ext_vector_type(16))) bf16  bf16x16;
typedef __attribute__((ext_vector_type(8)))  float f32x8;
typedef __attribute__((ext_vector_type(4)))  float f32x4;
typedef __attribute__((ext_vector_type(4)))  unsigned int u32x4;
typedef __attribute__((ext_vector_type(8)))  int   i32x8;
typedef __attribute__((ext_vector_type(4)))  int   i32x4;

#define D_MODEL 1024
#define NHEAD   16
#define KD      64
#define SEQ     2048
#define BATCH   2
#define MROWS   (BATCH * SEQ)   // 4096
#define SROW    2052            // padded LDS score row stride (floats)
#define LDSROW  72              // W-tile LDS row pitch (bf16 elems): 128B row + 16B TDM pad

// ---- WMMA fragment loaders --------------------------------------------------
// A fragment 16x32 bf16 (ISA 16-bit A table): lane l -> row (l&15),
// elements 0..7  <- K = 8*(l>>4) .. +7,  elements 8..15 <- K = 16+8*(l>>4) .. +7
__device__ __forceinline__ bf16x16 load_a_frag(const bf16* __restrict__ base, int ldm, int lane) {
  const bf16* p = base + (size_t)(lane & 15) * ldm + ((lane >> 4) << 3);
  bf16x8 lo = *(const bf16x8*)(p);
  bf16x8 hi = *(const bf16x8*)(p + 16);
  return __builtin_shufflevector(lo, hi, 0,1,2,3,4,5,6,7,8,9,10,11,12,13,14,15);
}

// B fragment 32x16 bf16 (same K order as A): lane l -> column (l&15),
// elements 0..7 <- K = 8*(l>>4) .. +7, elements 8..15 <- K = 16+8*(l>>4) .. +7.  Source stored row-major along K, leading dim ld.
__device__ __forceinline__ bf16x16 load_bT_frag(const bf16* base, int ld, int lane) {
  const bf16* p = base + (size_t)(lane & 15) * ld + ((lane >> 4) << 3);
  bf16x8 lo = *(const bf16x8*)(p);
  bf16x8 hi = *(const bf16x8*)(p + 16);
  return __builtin_shufflevector(lo, hi, 0,1,2,3,4,5,6,7,8,9,10,11,12,13,14,15);
}

// A fragment built from f32 data resident in LDS (convert on the fly).
__device__ __forceinline__ bf16x16 load_a_frag_lds_f32(const float* base, int ldm, int lane) {
  const float* p = base + (lane & 15) * ldm + ((lane >> 4) << 3);
  f32x4 a0 = *(const f32x4*)(p);
  f32x4 a1 = *(const f32x4*)(p + 4);
  f32x4 b0 = *(const f32x4*)(p + 16);
  f32x4 b1 = *(const f32x4*)(p + 20);
  bf16x16 f;
#pragma unroll
  for (int i = 0; i < 4; ++i) {
    f[i]      = (bf16)a0[i];
    f[i + 4]  = (bf16)a1[i];
    f[i + 8]  = (bf16)b0[i];
    f[i + 12] = (bf16)b1[i];
  }
  return f;
}

__device__ __forceinline__ f32x8 wmma_bf16(bf16x16 a, bf16x16 b, f32x8 c) {
  return __builtin_amdgcn_wmma_f32_16x16x32_bf16(false, a, false, b, (short)0, c, false, false);
}

// ---- TDM: DMA a 64(rows) x 64(k) bf16 tile of W into LDS --------------------
// Descriptor per CDNA5 ISA ch.8: group0 {count=1, lds_addr, global_addr, type=2},
// group1 {data_size=2B, pad 4 DW every 32 DW -> 144B LDS pitch, tensor 1024x1024,
// tile 64x64, stride0=1024}.  Groups 2/3 zero (2-D tile).
// 6-arg builtin form (clang-23 / therock-10.0 headers).
__device__ __forceinline__ void tdm_load_wtile(const bf16* gsrc, unsigned lds_off) {
  unsigned long long ga = (unsigned long long)(uintptr_t)gsrc;
  u32x4 g0;
  g0[0] = 1u;                                               // count=1, user descriptor
  g0[1] = lds_off;                                          // LDS byte address
  g0[2] = (unsigned)ga;                                     // global addr [31:0]
  g0[3] = (unsigned)((ga >> 32) & 0x01ffffffu) | (2u << 30);// global addr [56:32] | type=2
  i32x8 g1;
  g1[0] = (1 << 16)    // data_size = 2 bytes
        | (1 << 20)    // pad_enable
        | (4 << 22)    // pad_interval: 32 DWORDs (=128B tile row)
        | (3 << 25);   // pad_amount: 4 DWORDs (=16B -> 144B pitch)
  g1[1] = (int)(((unsigned)D_MODEL & 0xffffu) << 16);                    // tensor_dim0 lo16
  g1[2] = (int)(((unsigned)D_MODEL >> 16) |
                (((unsigned)D_MODEL & 0xffffu) << 16));                  // dim0 hi | dim1 lo
  g1[3] = (int)(((unsigned)D_MODEL >> 16) | (64u << 16));                // dim1 hi | tile_dim0=64
  g1[4] = (int)64u;                                                      // tile_dim1=64 rows
  g1[5] = (int)D_MODEL;                                                  // tensor_dim0_stride lo32
  g1[6] = 0;
  g1[7] = 0;
  i32x4 gz4 = {0, 0, 0, 0};
  i32x8 gz8 = {0, 0, 0, 0, 0, 0, 0, 0};
  __builtin_amdgcn_tensor_load_to_lds(g0, g1, gz4, gz4, gz8, 0);
}

// ---- f32 -> bf16 convert (vectorized x4) ------------------------------------
__global__ void cvt_f32_to_bf16(const float* __restrict__ src, bf16* __restrict__ dst, int n4) {
  int i = blockIdx.x * blockDim.x + threadIdx.x;
  const int stride = gridDim.x * blockDim.x;
  for (; i < n4; i += stride) {
    f32x4 v = ((const f32x4*)src)[i];
    bf16x4 o;
#pragma unroll
    for (int j = 0; j < 4; ++j) o[j] = (bf16)v[j];
    ((bf16x4*)dst)[i] = o;
  }
}

// ---- GEMM: C[m,n] = sum_k X[m,k] * W[n,k] + bias[n] -------------------------
// 128 threads = 4 waves. Wave w owns rows [m0+16w,+16); block owns 64 cols.
// W tile (64x64 bf16) is TDM-DMA'd to LDS (double buffered) by wave 0 and
// shared by all 4 waves; A fragments stream straight from global.
// mode 0: bf16 out [b,h,s,d]  (Q,K) | mode 1: bf16 out [b,h,d,s] (Vt)
// mode 2: f32 out row-major [m,D]   (final projection -> d_out)
__global__ __launch_bounds__(128) void gemm_xwt(const bf16* __restrict__ X,
                                                const bf16* __restrict__ W,
                                                const float* __restrict__ bias,
                                                void* __restrict__ dstv, int mode) {
  __shared__ __align__(16) bf16 wtile[2][64 * LDSROW];   // 2 x 9216 B

  const int lane = threadIdx.x & 31;
  const int wave = threadIdx.x >> 5;
  const int m0 = blockIdx.x * 64 + wave * 16;
  const int n0 = blockIdx.y * 64;

  const f32x8 zero = {0.f, 0.f, 0.f, 0.f, 0.f, 0.f, 0.f, 0.f};
  f32x8 acc[4];
#pragma unroll
  for (int nt = 0; nt < 4; ++nt) acc[nt] = zero;

  const bf16* a_ptr = X + (size_t)m0 * D_MODEL;
  const bf16* w_ptr = W + (size_t)n0 * D_MODEL;

  const int NKT = D_MODEL / 64;   // 16 k-steps of 64
  if (wave == 0) tdm_load_wtile(w_ptr, (unsigned)(uintptr_t)&wtile[0][0]);

  for (int kt = 0; kt < NKT; ++kt) {
    if (wave == 0) {
      if (kt + 1 < NKT) {
        tdm_load_wtile(w_ptr + (size_t)(kt + 1) * 64,
                       (unsigned)(uintptr_t)&wtile[(kt + 1) & 1][0]);
        __builtin_amdgcn_s_wait_tensorcnt(1);   // current tile landed
      } else {
        __builtin_amdgcn_s_wait_tensorcnt(0);
      }
    }
    __syncthreads();                            // publish tile kt

    const int k = kt * 64;
    if (kt + 1 < NKT)
      __builtin_prefetch(a_ptr + (size_t)(lane & 15) * D_MODEL + k + 64, 0, 1);
    bf16x16 a0 = load_a_frag(a_ptr + k,      D_MODEL, lane);
    bf16x16 a1 = load_a_frag(a_ptr + k + 32, D_MODEL, lane);

    const bf16* bt = &wtile[kt & 1][0];
#pragma unroll
    for (int nt = 0; nt < 4; ++nt) {
      bf16x16 b0 = load_bT_frag(bt + nt * 16 * LDSROW,      LDSROW, lane);
      bf16x16 b1 = load_bT_frag(bt + nt * 16 * LDSROW + 32, LDSROW, lane);
      acc[nt] = wmma_bf16(a0, b0, acc[nt]);
      acc[nt] = wmma_bf16(a1, b1, acc[nt]);
    }
    __syncthreads();                            // tile kt fully consumed
  }

  const int col_lo = lane & 15;
  const int rhalf  = lane >> 4;
#pragma unroll
  for (int nt = 0; nt < 4; ++nt) {
    const int n = n0 + nt * 16 + col_lo;
    const float bv = bias ? bias[n] : 0.0f;
#pragma unroll
    for (int r = 0; r < 8; ++r) {
      const int m = m0 + r + 8 * rhalf;
      const float v = acc[nt][r] + bv;
      if (mode == 2) {
        ((float*)dstv)[(size_t)m * D_MODEL + n] = v;
      } else {
        const int bb = m >> 11, s = m & (SEQ - 1);
        const int h  = n >> 6,  d = n & (KD - 1);
        bf16* dst = (bf16*)dstv;
        if (mode == 0)
          dst[(((size_t)bb * NHEAD + h) * SEQ + s) * KD + d] = (bf16)v;
        else
          dst[(((size_t)bb * NHEAD + h) * KD + d) * SEQ + s] = (bf16)v;
      }
    }
  }
}

// ---- Fused attention: scores -> softmax (in LDS) -> attn store -> P@V ------
// One workgroup (4 waves) per (b, h, 16 query rows). Score block 16 x 2048 f32
// lives entirely in LDS (~128 KB of the WGP's 320 KB), so attn is written to
// HBM exactly once and P@V reads probabilities straight from LDS.
__global__ __launch_bounds__(128) void attn_fused(const bf16* __restrict__ Q,
                                                  const bf16* __restrict__ Kmat,
                                                  const bf16* __restrict__ Vt,
                                                  float* __restrict__ attn,
                                                  bf16* __restrict__ O) {
  extern __shared__ char smem_raw[];
  float* Sc    = (float*)smem_raw;                              // 16 x SROW
  float* red   = (float*)(smem_raw + 16 * SROW * 4);            // 128 partials
  float* rstat = red + 128;                                     // [0..15]=max, [16..31]=1/sum

  const int tid  = threadIdx.x;
  const int lane = tid & 31;
  const int wave = tid >> 5;
  const int q0 = blockIdx.x * 16;
  const int h  = blockIdx.y;
  const int b  = blockIdx.z;
  const size_t bh = (size_t)b * NHEAD + h;

  const bf16* Qbh = Q    + (bh * SEQ + q0) * KD;
  const bf16* Kbh = Kmat +  bh * SEQ * KD;
  const bf16* Vbh = Vt   +  bh * KD * SEQ;
  float* attn_base = attn + bh * (size_t)SEQ * SEQ + (size_t)q0 * SEQ;

  // Phase 1: scores = (Q Kt) / sqrt(64); each wave takes every 4th 16x16 tile.
  bf16x16 qa0 = load_a_frag(Qbh,      KD, lane);
  bf16x16 qa1 = load_a_frag(Qbh + 32, KD, lane);
  for (int jt = wave; jt < SEQ / 16; jt += 4) {
    const bf16* kb = Kbh + (size_t)jt * 16 * KD;
    bf16x16 b0 = load_bT_frag(kb,      KD, lane);
    bf16x16 b1 = load_bT_frag(kb + 32, KD, lane);
    f32x8 acc = {0.f, 0.f, 0.f, 0.f, 0.f, 0.f, 0.f, 0.f};
    acc = wmma_bf16(qa0, b0, acc);
    acc = wmma_bf16(qa1, b1, acc);
    const int col = jt * 16 + (lane & 15);
#pragma unroll
    for (int r = 0; r < 8; ++r)
      Sc[(r + 8 * (lane >> 4)) * SROW + col] = acc[r] * 0.125f;
  }
  __syncthreads();

  // Phase 2: row softmax stats (8 threads per row).
  const int m = tid >> 3, j = tid & 7;
  float lmax = -3.402823e38f;
  for (int i = 0; i < SEQ / 8; ++i)
    lmax = fmaxf(lmax, Sc[m * SROW + j + 8 * i]);
  red[m * 8 + j] = lmax;
  __syncthreads();
  if (tid < 16) {
    float mx = red[tid * 8];
#pragma unroll
    for (int i = 1; i < 8; ++i) mx = fmaxf(mx, red[tid * 8 + i]);
    rstat[tid] = mx;
  }
  __syncthreads();
  const float rmax = rstat[m];
  float lsum = 0.f;
  for (int i = 0; i < SEQ / 8; ++i) {
    const int idx = m * SROW + j + 8 * i;
    const float e = __expf(Sc[idx] - rmax);
    Sc[idx] = e;
    lsum += e;
  }
  red[m * 8 + j] = lsum;
  __syncthreads();
  if (tid < 16) {
    float s = 0.f;
#pragma unroll
    for (int i = 0; i < 8; ++i) s += red[tid * 8 + i];
    rstat[16 + tid] = 1.0f / s;
  }
  __syncthreads();

  // Phase 3: normalize in LDS + single coalesced attn store to HBM.
  for (int idx = tid; idx < 16 * SEQ; idx += 128) {
    const int mm = idx >> 11, c = idx & (SEQ - 1);
    const float p = Sc[mm * SROW + c] * rstat[16 + mm];
    Sc[mm * SROW + c] = p;
    attn_base[(size_t)mm * SEQ + c] = p;
  }
  __syncthreads();

  // Phase 4: O(16x64) = P(16x2048) @ V(2048x64); wave w owns d-columns [16w,+16).
  f32x8 oacc = {0.f, 0.f, 0.f, 0.f, 0.f, 0.f, 0.f, 0.f};
  const bf16* vbase = Vbh + (size_t)(wave * 16) * SEQ;   // Vt rows are d, contiguous in k
  for (int kk = 0; kk < SEQ; kk += 32) {
    bf16x16 pa = load_a_frag_lds_f32(Sc + kk, SROW, lane);
    bf16x16 vb = load_bT_frag(vbase + kk, SEQ, lane);
    oacc = wmma_bf16(pa, vb, oacc);
  }
  {
    const int ncol = lane & 15, rh = lane >> 4;
    bf16* obase = O + ((size_t)b * SEQ + q0) * D_MODEL + h * KD + wave * 16 + ncol;
#pragma unroll
    for (int r = 0; r < 8; ++r)
      obase[(size_t)(r + 8 * rh) * D_MODEL] = (bf16)oacc[r];
  }
}

// ---------------------------------------------------------------------------
extern "C" void kernel_launch(void* const* d_in, const int* in_sizes, int n_in,
                              void* d_out, int out_size, void* d_ws, size_t ws_size,
                              hipStream_t stream) {
  const float* q_in = (const float*)d_in[0];
  const float* k_in = (const float*)d_in[1];
  const float* v_in = (const float*)d_in[2];
  const float* Wq = (const float*)d_in[3];  const float* bq = (const float*)d_in[4];
  const float* Wk = (const float*)d_in[5];  const float* bk = (const float*)d_in[6];
  const float* Wv = (const float*)d_in[7];  const float* bv = (const float*)d_in[8];
  const float* Wo = (const float*)d_in[9];  const float* bo = (const float*)d_in[10];

  const size_t NX = (size_t)MROWS * D_MODEL;     // 4M activation elements
  const size_t NW = (size_t)D_MODEL * D_MODEL;   // 1M weight elements

  bf16* Xq  = (bf16*)d_ws;
  bf16* Xk  = Xq  + NX;
  bf16* Xv  = Xk  + NX;
  bf16* Wqb = Xv  + NX;
  bf16* Wkb = Wqb + NW;
  bf16* Wvb = Wkb + NW;
  bf16* Wob = Wvb + NW;
  bf16* Qb  = Wob + NW;   // [b,h,s,d]
  bf16* Kb  = Qb  + NX;   // [b,h,s,d]
  bf16* Vtb = Kb  + NX;   // [b,h,d,s]
  bf16* Ob  = Vtb + NX;   // [m, D] row-major

  float* out_p  = (float*)d_out;
  float* attn_p = out_p + NX;

  // 1) f32 -> bf16 converts (x4 vectorized)
  cvt_f32_to_bf16<<<1024, 256, 0, stream>>>(q_in, Xq, (int)(NX / 4));
  cvt_f32_to_bf16<<<1024, 256, 0, stream>>>(k_in, Xk, (int)(NX / 4));
  cvt_f32_to_bf16<<<1024, 256, 0, stream>>>(v_in, Xv, (int)(NX / 4));
  cvt_f32_to_bf16<<<256, 256, 0, stream>>>(Wq, Wqb, (int)(NW / 4));
  cvt_f32_to_bf16<<<256, 256, 0, stream>>>(Wk, Wkb, (int)(NW / 4));
  cvt_f32_to_bf16<<<256, 256, 0, stream>>>(Wv, Wvb, (int)(NW / 4));
  cvt_f32_to_bf16<<<256, 256, 0, stream>>>(Wo, Wob, (int)(NW / 4));

  // 2) QKV projections (WMMA + TDM-staged W tiles). V stored transposed per head.
  dim3 gg(MROWS / 64, D_MODEL / 64);
  gemm_xwt<<<gg, 128, 0, stream>>>(Xq, Wqb, bq, Qb, 0);
  gemm_xwt<<<gg, 128, 0, stream>>>(Xk, Wkb, bk, Kb, 0);
  gemm_xwt<<<gg, 128, 0, stream>>>(Xv, Wvb, bv, Vtb, 1);

  // 3) Fused attention (131 KB dynamic LDS per workgroup).
  const size_t smem = (size_t)16 * SROW * 4 + 128 * 4 + 32 * 4;
  (void)hipFuncSetAttribute(reinterpret_cast<const void*>(attn_fused),
                            hipFuncAttributeMaxDynamicSharedMemorySize, (int)smem);
  attn_fused<<<dim3(SEQ / 16, NHEAD, BATCH), 128, smem, stream>>>(Qb, Kb, Vtb, attn_p, Ob);

  // 4) Output projection -> f32 d_out
  gemm_xwt<<<gg, 128, 0, stream>>>(Ob, Wob, bo, out_p, 2);
}
